// mha_stack_25941602468145
// MI455X (gfx1250) — matrix operand order repaired and checked by reading, not compiled, not run on hardware
//
#include <hip/hip_runtime.h>
#include <hip/hip_bf16.h>

// ---------------------------------------------------------------------------
// CDNA5 (gfx1250) bf16 WMMA implementation of the 3-layer MHA encoder stack.
// wave32; V_WMMA_F32_16X16X32_BF16 for all GEMMs (projections, QK^T, PV, FF).
// GEMM A-tiles staged by the Tensor Data Mover (TENSOR_LOAD_TO_LDS, TENSORcnt),
// falling back to GLOBAL_LOAD_ASYNC_TO_LDS_B128 if the TDM builtin is absent.
// ---------------------------------------------------------------------------

typedef __bf16 bf16;
typedef bf16  v16bf __attribute__((ext_vector_type(16)));
typedef bf16  v8bf  __attribute__((ext_vector_type(8)));
typedef float v8f   __attribute__((ext_vector_type(8)));
typedef unsigned int v4u __attribute__((ext_vector_type(4)));
typedef int v8i __attribute__((ext_vector_type(8)));
typedef int v4i __attribute__((ext_vector_type(4)));

union Frag16 { v16bf v; v8bf h[2]; };

#define B_SZ    2
#define S_LEN   2048
#define DMODEL  1024
#define NHEAD   4
#define DHEAD   256
#define MROWS   (B_SZ * S_LEN)      // 4096
#define NEGBIG  (-4294967295.0f)

#if defined(__has_builtin)
#  if __has_builtin(__builtin_amdgcn_tensor_load_to_lds)
#    define HAVE_TDM 1
#  else
#    define HAVE_TDM 0
#  endif
#else
#  define HAVE_TDM 0
#endif

static __device__ __forceinline__ v8f wmma_bf16(const Frag16& a, const Frag16& b, v8f c) {
  return __builtin_amdgcn_wmma_f32_16x16x32_bf16(false, a.v, false, b.v, (short)0, c, false, false);
}

// Async global->LDS DMA (VGPR-free), tracked by ASYNCcnt. (fallback path)
static __device__ __forceinline__ void async_load_b128(const void* gaddr, void* lds) {
  unsigned ldsOff = (unsigned)(uintptr_t)lds;           // LDS aperture: low 32 bits
  asm volatile("global_load_async_to_lds_b128 %0, %1, off"
               :: "v"(ldsOff), "v"((unsigned long long)(uintptr_t)gaddr)
               : "memory");
}
static __device__ __forceinline__ void wait_async0() {
  asm volatile("s_wait_asynccnt 0x0" ::: "memory");
}

#define GT_MB 128
#define GT_NB 128
#define GT_K  32
#define BT_STRIDE 40

#if HAVE_TDM
// TDM: DMA one 2D tile (GT_MB rows x GT_K cols, bf16) of a row-major
// [Mdim x Kdim] tensor into contiguous LDS. D# bitfields per CDNA5 ISA ch.8.
static __device__ __forceinline__ void tdm_load_tile(const bf16* gaddr, void* lds,
                                                     int Mdim, int Kdim) {
  unsigned long long ga = (unsigned long long)(uintptr_t)gaddr;
  v4u g0;
  g0.x = 1u;                                            // count=1, user desc
  g0.y = (unsigned)(uintptr_t)lds;                      // lds_addr (bytes)
  g0.z = (unsigned)(ga & 0xffffffffu);                  // global_addr[31:0]
  g0.w = (unsigned)((ga >> 32) & 0x1ffffffu) | (2u << 30);  // addr[56:32] | type=2
  v8i g1;
  g1[0] = (int)(1u << 16);                              // data_size=1 (2 bytes)
  g1[1] = (int)(((unsigned)Kdim & 0xffffu) << 16);      // tensor_dim0[15:0]
  g1[2] = (int)((((unsigned)Kdim >> 16) & 0xffffu) |    // tensor_dim0[31:16]
                (((unsigned)Mdim & 0xffffu) << 16));    // tensor_dim1[15:0]
  g1[3] = (int)((((unsigned)Mdim >> 16) & 0xffffu) |    // tensor_dim1[31:16]
                ((unsigned)GT_K << 16));                // tile_dim0 = 32
  g1[4] = GT_MB;                                        // tile_dim1 = 128
  g1[5] = Kdim;                                         // tensor_dim0_stride[31:0]
  g1[6] = 0;
  g1[7] = 0;
  v4i gz = {0, 0, 0, 0};
#if __clang_major__ >= 23
  v8i gz8 = {0, 0, 0, 0, 0, 0, 0, 0};
  __builtin_amdgcn_tensor_load_to_lds(g0, g1, gz, gz, gz8, 0);
#else
  __builtin_amdgcn_tensor_load_to_lds(g0, g1, gz, gz, 0);
#endif
}
#endif

// ---------------------------------------------------------------- convert ---
__global__ __launch_bounds__(256) void cvt_f32_bf16_kernel(const float* __restrict__ x,
                                                           bf16* __restrict__ y, int n) {
  int i = (blockIdx.x * 256 + threadIdx.x) * 4;
  if (i + 3 < n) {
    float4 v = *(const float4*)(x + i);
    y[i + 0] = (bf16)v.x; y[i + 1] = (bf16)v.y;
    y[i + 2] = (bf16)v.z; y[i + 3] = (bf16)v.w;
  }
}

// ------------------------------------------------------------- row masks ----
__global__ __launch_bounds__(256) void row_mask_kernel(const float* __restrict__ x,
                                                       float* __restrict__ mask) {
  __shared__ float red[256];
  int row = blockIdx.x, tid = threadIdx.x;
  const float* p = x + (size_t)row * DMODEL + tid * 4;
  float s = p[0] + p[1] + p[2] + p[3];
  red[tid] = s; __syncthreads();
  for (int st = 128; st > 0; st >>= 1) {
    if (tid < st) red[tid] += red[tid + st];
    __syncthreads();
  }
  if (tid == 0) mask[row] = (red[0] != 0.0f) ? 1.0f : 0.0f;
}

// ------------------------------------------------------------------ GEMM ----
// C = epilogue(A[M,K] @ W[K,N] + bias[N])
// MODE 0: relu -> outB (bf16)
// MODE 1: leaky_relu(0.2) + res -> outF (f32) and outB (bf16)
// Block tile 128x128, K-step 32, 8 waves, wave tile 64x32 (8 WMMA / K-step).
// Double-buffered LDS; A tile staged by TDM DMA (wave 0 issues, TENSORcnt
// gated), W tile transposed through registers.
template <int MODE>
__global__ __launch_bounds__(256) void gemm_bf16_kernel(
    const bf16* __restrict__ A, const bf16* __restrict__ W,
    const float* __restrict__ bias, const float* __restrict__ res,
    bf16* __restrict__ outB, float* __restrict__ outF,
    int Mdim, int Ndim, int Kdim) {
  __shared__ bf16 As[2][GT_MB * GT_K];        // 2 x 8 KB
  __shared__ bf16 Bt[2][GT_NB * BT_STRIDE];   // 2 x 10 KB, transposed [n][k]

  const int tid  = threadIdx.x;
  const int w    = tid >> 5;
  const int lane = tid & 31;
  const int half = lane >> 4;
  const int l16  = lane & 15;
  const int wmBase = (w & 1) * 64;            // wave M offset in block tile
  const int wnBase = (w >> 1) * 32;           // wave N offset in block tile
  const int mBase = blockIdx.y * GT_MB;
  const int nBase = blockIdx.x * GT_NB;

  // staging chunk coordinates (8-half chunks)
  const int aRow0 = tid >> 2,           aCc0 = (tid & 3) * 8;
  const int aRow1 = (tid + 256) >> 2,   aCc1 = ((tid + 256) & 3) * 8;
  const int wKk0  = tid >> 4,           wCc0 = (tid & 15) * 8;
  const int wKk1  = (tid + 256) >> 4,   wCc1 = ((tid + 256) & 15) * 8;
  (void)aRow0; (void)aCc0; (void)aRow1; (void)aCc1;

  v8f acc[4][2];
  #pragma unroll
  for (int a = 0; a < 4; ++a)
    #pragma unroll
    for (int b = 0; b < 2; ++b)
      #pragma unroll
      for (int e = 0; e < 8; ++e) acc[a][b][e] = 0.0f;

  // ---- prologue: stage k-tile 0 into buffer 0
  {
#if HAVE_TDM
    if (tid < 32)
      tdm_load_tile(A + (size_t)mBase * Kdim, &As[0][0], Mdim, Kdim);
#else
    async_load_b128(A + (size_t)(mBase + aRow0) * Kdim + aCc0, &As[0][aRow0 * GT_K + aCc0]);
    async_load_b128(A + (size_t)(mBase + aRow1) * Kdim + aCc1, &As[0][aRow1 * GT_K + aCc1]);
#endif
    v8bf gw0 = *(const v8bf*)(W + (size_t)wKk0 * Ndim + nBase + wCc0);
    v8bf gw1 = *(const v8bf*)(W + (size_t)wKk1 * Ndim + nBase + wCc1);
    #pragma unroll
    for (int e = 0; e < 8; ++e) {
      Bt[0][(wCc0 + e) * BT_STRIDE + wKk0] = gw0[e];
      Bt[0][(wCc1 + e) * BT_STRIDE + wKk1] = gw1[e];
    }
#if HAVE_TDM
    if (tid < 32) __builtin_amdgcn_s_wait_tensorcnt(0);
#else
    wait_async0();
#endif
  }

  const int nk = Kdim / GT_K;
  for (int kt = 0; kt < nk; ++kt) {
    const int buf = kt & 1;
    __syncthreads();

    // ---- issue next tile's loads (overlap with compute)
    v8bf gw0, gw1;
    const bool have = (kt + 1) < nk;
    if (have) {
      const int k0 = (kt + 1) * GT_K;
      __builtin_prefetch(W + (size_t)(k0 + GT_K) * Ndim + nBase, 0, 1);
#if HAVE_TDM
      if (tid < 32)
        tdm_load_tile(A + (size_t)mBase * Kdim + k0, &As[buf ^ 1][0], Mdim, Kdim);
#else
      async_load_b128(A + (size_t)(mBase + aRow0) * Kdim + k0 + aCc0,
                      &As[buf ^ 1][aRow0 * GT_K + aCc0]);
      async_load_b128(A + (size_t)(mBase + aRow1) * Kdim + k0 + aCc1,
                      &As[buf ^ 1][aRow1 * GT_K + aCc1]);
#endif
      gw0 = *(const v8bf*)(W + (size_t)(k0 + wKk0) * Ndim + nBase + wCc0);
      gw1 = *(const v8bf*)(W + (size_t)(k0 + wKk1) * Ndim + nBase + wCc1);
    }

    // ---- compute on current buffer
    Frag16 af[4];
    #pragma unroll
    for (int ms = 0; ms < 4; ++ms) {
      const bf16* ap = &As[buf][(wmBase + ms * 16 + l16) * GT_K + half * 8];
      af[ms].h[0] = *(const v8bf*)ap;
      af[ms].h[1] = *(const v8bf*)(ap + 16);
    }
    #pragma unroll
    for (int ns = 0; ns < 2; ++ns) {
      Frag16 bfg;
      const bf16* bp = &Bt[buf][(wnBase + ns * 16 + l16) * BT_STRIDE + half * 8];
      bfg.h[0] = *(const v8bf*)bp;
      bfg.h[1] = *(const v8bf*)(bp + 16);
      #pragma unroll
      for (int ms = 0; ms < 4; ++ms)
        acc[ms][ns] = wmma_bf16(af[ms], bfg, acc[ms][ns]);
    }

    // ---- finish staging next buffer
    if (have) {
      #pragma unroll
      for (int e = 0; e < 8; ++e) {
        Bt[buf ^ 1][(wCc0 + e) * BT_STRIDE + wKk0] = gw0[e];
        Bt[buf ^ 1][(wCc1 + e) * BT_STRIDE + wKk1] = gw1[e];
      }
#if HAVE_TDM
      if (tid < 32) __builtin_amdgcn_s_wait_tensorcnt(0);
#else
      wait_async0();
#endif
    }
  }

  // ---- epilogue (straight-line; MODE is compile-time)
  #pragma unroll
  for (int ms = 0; ms < 4; ++ms) {
    #pragma unroll
    for (int ns = 0; ns < 2; ++ns) {
      int n = nBase + wnBase + ns * 16 + l16;
      float bval = bias[n];
      #pragma unroll
      for (int r = 0; r < 8; ++r) {
        int m = mBase + wmBase + ms * 16 + r + half * 8;
        size_t idx = (size_t)m * Ndim + n;
        float v = acc[ms][ns][r] + bval;
        if (MODE == 0) {
          v = fmaxf(v, 0.0f);
          outB[idx] = (bf16)v;
        } else {
          float y = (v > 0.0f) ? v : 0.2f * v;
          y += res[idx];
          outF[idx] = y;
          outB[idx] = (bf16)y;
        }
      }
    }
  }
}

// ----------------------------------------------------------- attention ------
// One block = (b, h, 64 queries); 4 waves, each wave owns 16 queries.
// Flash-style online softmax over 32-key tiles; all matmuls via WMMA bf16.
#define KT 32
#define VT_STRIDE 40
#define P_STRIDE  40

__global__ __launch_bounds__(128) void flash_attn_kernel(
    const bf16* __restrict__ Qp, const bf16* __restrict__ Kp,
    const bf16* __restrict__ Vp, const float* __restrict__ kmask,
    const float* __restrict__ qmask, float* __restrict__ O) {
  __shared__ bf16 Vt[DHEAD * VT_STRIDE];     // 20 KB, transposed [dh][key]
  __shared__ bf16 Psh[4][16 * P_STRIDE];     // 5 KB, per-wave P tile

  const int b    = blockIdx.z;
  const int h    = blockIdx.y;
  const int tid  = threadIdx.x;
  const int w    = tid >> 5;
  const int lane = tid & 31;
  const int half = lane >> 4;
  const int l16  = lane & 15;
  const int rowBase = b * S_LEN;
  const int q0 = blockIdx.x * 64 + w * 16;

  // Q A-fragments for this wave's 16 queries (8 k-chunks over dh=256)
  Frag16 qf[8];
  {
    const size_t qrow = (size_t)(rowBase + q0 + l16) * DMODEL + h * DHEAD;
    #pragma unroll
    for (int kc = 0; kc < 8; ++kc) {
      const bf16* ptr = Qp + qrow + kc * 32 + half * 8;
      qf[kc].h[0] = *(const v8bf*)ptr;
      qf[kc].h[1] = *(const v8bf*)(ptr + 16);
    }
  }

  v8f acc[16];
  for (int nf = 0; nf < 16; ++nf)
    for (int e = 0; e < 8; ++e) acc[nf][e] = 0.0f;
  float mrow[8], lrow[8];
  for (int r = 0; r < 8; ++r) { mrow[r] = -__builtin_inff(); lrow[r] = 0.0f; }

  for (int kt = 0; kt < S_LEN / KT; ++kt) {
    const int k0 = kt * KT;
    __syncthreads();
    // stage V tile transposed: Vt[dh][key], 32 keys x 256 dh
    #pragma unroll
    for (int j = 0; j < 8; ++j) {
      int c = tid + j * 128;
      int kk = c >> 5;
      int cc0 = (c & 31) * 8;
      v8bf vv = *(const v8bf*)(Vp + (size_t)(rowBase + k0 + kk) * DMODEL +
                               h * DHEAD + cc0);
      #pragma unroll
      for (int e = 0; e < 8; ++e) Vt[(cc0 + e) * VT_STRIDE + kk] = vv[e];
    }
    __syncthreads();

    // scores: Q(16x256) . K^T -> two 16x16 sub-tiles over 32 keys
    v8f s[2];
    for (int e = 0; e < 8; ++e) { s[0][e] = 0.0f; s[1][e] = 0.0f; }
    #pragma unroll
    for (int sub = 0; sub < 2; ++sub) {
      const size_t krow =
          (size_t)(rowBase + k0 + sub * 16 + l16) * DMODEL + h * DHEAD;
      #pragma unroll
      for (int kc = 0; kc < 8; ++kc) {
        Frag16 bfg;                                   // B-frag: K rows contiguous
        const bf16* ptr = Kp + krow + kc * 32 + half * 8;
        bfg.h[0] = *(const v8bf*)ptr;
        bfg.h[1] = *(const v8bf*)(ptr + 16);
        s[sub] = wmma_bf16(qf[kc], bfg, s[sub]);
      }
    }

    const float km0 = kmask[rowBase + k0 + l16];
    const float km1 = kmask[rowBase + k0 + 16 + l16];
    const float scale = 0.0625f;   // 1/sqrt(256)

    float rescale[8];
    #pragma unroll
    for (int r = 0; r < 8; ++r) {
      float a0 = (km0 == 0.0f) ? NEGBIG : s[0][r] * scale;
      float a1 = (km1 == 0.0f) ? NEGBIG : s[1][r] * scale;
      float mx = fmaxf(a0, a1);
      #pragma unroll
      for (int d = 1; d < 16; d <<= 1) mx = fmaxf(mx, __shfl_xor(mx, d, 32));
      float mnew = fmaxf(mrow[r], mx);
      rescale[r] = __expf(mrow[r] - mnew);
      float p0 = __expf(a0 - mnew);
      float p1 = __expf(a1 - mnew);
      s[0][r] = p0; s[1][r] = p1;
      float ps = p0 + p1;
      #pragma unroll
      for (int d = 1; d < 16; d <<= 1) ps += __shfl_xor(ps, d, 32);
      lrow[r] = lrow[r] * rescale[r] + ps;
      mrow[r] = mnew;
    }
    #pragma unroll
    for (int nf = 0; nf < 16; ++nf)
      #pragma unroll
      for (int r = 0; r < 8; ++r) acc[nf][r] *= rescale[r];

    // P: D-layout -> LDS (bf16) -> reload as A-fragment
    bf16* P = Psh[w];
    #pragma unroll
    for (int r = 0; r < 8; ++r) {
      int m = r + half * 8;
      P[m * P_STRIDE + l16]      = (bf16)s[0][r];
      P[m * P_STRIDE + 16 + l16] = (bf16)s[1][r];
    }
    Frag16 pf;
    {
      const bf16* pp = P + l16 * P_STRIDE + half * 8;
      pf.h[0] = *(const v8bf*)pp;
      pf.h[1] = *(const v8bf*)(pp + 16);
    }
    // out += P(16x32) x V(32x256)
    #pragma unroll
    for (int nf = 0; nf < 16; ++nf) {
      Frag16 vb;
      const bf16* vp = Vt + (nf * 16 + l16) * VT_STRIDE + half * 8;
      vb.h[0] = *(const v8bf*)vp;
      vb.h[1] = *(const v8bf*)(vp + 16);
      acc[nf] = wmma_bf16(pf, vb, acc[nf]);
    }
  }

  // epilogue: normalize, apply query mask, write fp32
  #pragma unroll
  for (int r = 0; r < 8; ++r) {
    int m = q0 + r + half * 8;
    float qm = qmask[rowBase + m];
    float inv = qm / lrow[r];
    size_t orow = (size_t)(rowBase + m) * DMODEL + h * DHEAD;
    #pragma unroll
    for (int nf = 0; nf < 16; ++nf)
      O[orow + nf * 16 + l16] = acc[nf][r] * inv;
  }
}

// --------------------------------------------------- residual + layernorm ---
__global__ __launch_bounds__(256) void ln_res_kernel(
    const float* __restrict__ X, const float* __restrict__ R,
    const float* __restrict__ g, const float* __restrict__ be,
    float* __restrict__ outF, bf16* __restrict__ outB) {
  __shared__ float red[256];
  int row = blockIdx.x, tid = threadIdx.x;
  const size_t base = (size_t)row * DMODEL;
  float v[4];
  float s = 0.0f;
  #pragma unroll
  for (int j = 0; j < 4; ++j) {
    int c = tid * 4 + j;
    v[j] = X[base + c] + R[base + c];
    s += v[j];
  }
  red[tid] = s; __syncthreads();
  for (int st = 128; st > 0; st >>= 1) {
    if (tid < st) red[tid] += red[tid + st];
    __syncthreads();
  }
  float mean = red[0] / DMODEL;
  __syncthreads();
  float sq = 0.0f;
  #pragma unroll
  for (int j = 0; j < 4; ++j) { float d = v[j] - mean; sq += d * d; }
  red[tid] = sq; __syncthreads();
  for (int st = 128; st > 0; st >>= 1) {
    if (tid < st) red[tid] += red[tid + st];
    __syncthreads();
  }
  float var = red[0] / (DMODEL - 1);            // unbiased (ddof=1)
  float istd = 1.0f / (sqrtf(var) + 1e-8f);
  #pragma unroll
  for (int j = 0; j < 4; ++j) {
    int c = tid * 4 + j;
    float y = g[c] * (v[j] - mean) * istd + be[c];
    outF[base + c] = y;
    outB[base + c] = (bf16)y;
  }
}

// ---------------------------------------------------------------- launch ----
extern "C" void kernel_launch(void* const* d_in, const int* in_sizes, int n_in,
                              void* d_out, int out_size, void* d_ws, size_t ws_size,
                              hipStream_t stream) {
  const float* queries = (const float*)d_in[0];
  const float* keysp   = (const float*)d_in[1];
  const float* values  = (const float*)d_in[2];
  const float* Wq = (const float*)d_in[3];
  const float* bq = (const float*)d_in[4];
  const float* Wk = (const float*)d_in[5];
  const float* bk = (const float*)d_in[6];
  const float* Wv = (const float*)d_in[7];
  const float* bv = (const float*)d_in[8];
  const float* gamma = (const float*)d_in[9];
  const float* beta  = (const float*)d_in[10];
  const float* Wff = (const float*)d_in[11];
  const float* bff = (const float*)d_in[12];

  const size_t MD = (size_t)MROWS * DMODEL;      // 4M elements
  const size_t WD = (size_t)DMODEL * DMODEL;     // 1M elements

  char* p = (char*)d_ws;
  auto carve = [&](size_t bytes) -> void* {
    void* r = (void*)p;
    p += (bytes + 255) & ~(size_t)255;
    return r;
  };
  bf16* in0q = (bf16*)carve(MD * 2);
  bf16* in0k = (bf16*)carve(MD * 2);
  bf16* in0v = (bf16*)carve(MD * 2);
  bf16* encB = (bf16*)carve(MD * 2);
  bf16* Pq   = (bf16*)carve(MD * 2);
  bf16* Pk   = (bf16*)carve(MD * 2);
  bf16* Pv   = (bf16*)carve(MD * 2);
  bf16* lnB  = (bf16*)carve(MD * 2);
  bf16* Wc   = (bf16*)carve(WD * 2);
  float* attO = (float*)carve(MD * 4);
  float* lnF  = (float*)carve(MD * 4);
  float* ffF  = (float*)carve(MD * 4);
  float* qm   = (float*)carve(MROWS * 4);
  float* km   = (float*)carve(MROWS * 4);

  const dim3 gemmGrid(DMODEL / GT_NB, MROWS / GT_MB);  // (8, 32)
  const dim3 attnGrid(S_LEN / 64, NHEAD, B_SZ);        // (32, 4, 2)

  // layer-0 bf16 activations
  cvt_f32_bf16_kernel<<<(int)(MD / 1024), 256, 0, stream>>>(queries, in0q, (int)MD);
  cvt_f32_bf16_kernel<<<(int)(MD / 1024), 256, 0, stream>>>(keysp,   in0k, (int)MD);
  cvt_f32_bf16_kernel<<<(int)(MD / 1024), 256, 0, stream>>>(values,  in0v, (int)MD);

  const float* qf = queries;
  const float* kf = keysp;
  const bf16 *aq = in0q, *ak = in0k, *av = in0v;

  for (int i = 0; i < 3; ++i) {
    row_mask_kernel<<<MROWS, 256, 0, stream>>>(qf, qm);
    row_mask_kernel<<<MROWS, 256, 0, stream>>>(kf, km);

    // Q = relu(aq @ Wq + bq)
    cvt_f32_bf16_kernel<<<(int)(WD / 1024), 256, 0, stream>>>(Wq + (size_t)i * WD, Wc, (int)WD);
    gemm_bf16_kernel<0><<<gemmGrid, 256, 0, stream>>>(aq, Wc, bq + i * DMODEL,
        nullptr, Pq, nullptr, MROWS, DMODEL, DMODEL);
    // K
    cvt_f32_bf16_kernel<<<(int)(WD / 1024), 256, 0, stream>>>(Wk + (size_t)i * WD, Wc, (int)WD);
    gemm_bf16_kernel<0><<<gemmGrid, 256, 0, stream>>>(ak, Wc, bk + i * DMODEL,
        nullptr, Pk, nullptr, MROWS, DMODEL, DMODEL);
    // V
    cvt_f32_bf16_kernel<<<(int)(WD / 1024), 256, 0, stream>>>(Wv + (size_t)i * WD, Wc, (int)WD);
    gemm_bf16_kernel<0><<<gemmGrid, 256, 0, stream>>>(av, Wc, bv + i * DMODEL,
        nullptr, Pv, nullptr, MROWS, DMODEL, DMODEL);

    flash_attn_kernel<<<attnGrid, 128, 0, stream>>>(Pq, Pk, Pv, km, qm, attO);

    ln_res_kernel<<<MROWS, 256, 0, stream>>>(attO, qf, gamma + i * DMODEL,
                                             beta + i * DMODEL, lnF, lnB);

    // FF: leaky_relu(lnF @ Wff + bff) + lnF
    cvt_f32_bf16_kernel<<<(int)(WD / 1024), 256, 0, stream>>>(Wff + (size_t)i * WD, Wc, (int)WD);
    float* outF = (i == 2) ? (float*)d_out : ffF;
    gemm_bf16_kernel<1><<<gemmGrid, 256, 0, stream>>>(lnB, Wc, bff + i * DMODEL,
        lnF, encB, outF, MROWS, DMODEL, DMODEL);

    qf = outF; kf = outF;
    aq = encB; ak = encB; av = encB;
  }
}
